// GLSGRFeaturePyramidNetwork_24910810316993
// MI455X (gfx1250) — matrix operand order repaired and checked by reading, not compiled, not run on hardware
//
#include <hip/hip_runtime.h>

// GLSGR-FPN for gfx1250 (MI455X). All GEMM-shaped math (patch conv, fc2, gram,
// GATv2 linear transforms, fc3, convT, 1x1 inner convs, 3x3 layer convs as 9
// shifted-GEMM taps) runs through v_wmma_f32_16x16x32_bf16 with f32 accumulate.
// Activations are staged channels-last bf16 so convs are implicit GEMMs.
// Assumptions: k == 7 (reference default; k arrives as a device scalar and
// cannot be read during graph capture); ws_size >= ~220 MB.
//
// v2: branchless vectorized tile staging (one b128 global load per thread per
// K-step for each of A and B, validity folded into precomputed clamps/masks).

typedef __attribute__((ext_vector_type(16))) __bf16 v16bf;
typedef __attribute__((ext_vector_type(8)))  float  v8f;
typedef unsigned short u16;
typedef unsigned int   u32;

__device__ __forceinline__ u16 f2bf(float f) {
  u32 u = __float_as_uint(f);
  u32 lsb = (u >> 16) & 1u;
  u += 0x7fffu + lsb;                 // round-to-nearest-even
  return (u16)(u >> 16);
}

// ---------------------------------------------------------------------------
// Generic bf16 WMMA GEMM: C[M,N] = op(A[M,K] @ B[K,N]).
// A may be addressed in "shifted image" mode (img_h>0): row m -> (h,w) in an
// [img_h, img_w, K] channels-last image, sampled at (h+sdy, w+sdx), zero pad.
// Epilogue: v = acc + bias[n]; if(relu) v=max(v,0); v += Cin[m*N+n];
// outputs: f32 Cout[m*N+n] and/or bf16 Obf[m*obf_stride + obf_coloff + n].
// Requirements: K % 32 == 0, N % 8 == 0, A/B 16B-aligned, B may be over-read
// by <128B (workspace slack), OOB M rows clamped (outputs store-guarded).
// ---------------------------------------------------------------------------
__global__ __launch_bounds__(256) void gemm_wmma_bf16(
    const u16* __restrict__ A, const u16* __restrict__ B,
    const float* __restrict__ Cin, const float* __restrict__ bias,
    float* __restrict__ Cout, u16* __restrict__ Obf,
    int M, int N, int K,
    int img_w, int img_h, int sdy, int sdx,
    int relu, int obf_stride, int obf_coloff)
{
  __shared__ alignas(16) u16 lA[64][40];   // 64x32 tile; 80B pitch (16B mult)
  __shared__ alignas(16) u16 lB[32][72];   // 32x64 tile; 144B pitch (16B mult)
  const int t    = threadIdx.x;
  const int lane = t & 31;
  const int wv   = t >> 5;
  const int wm   = wv & 3;           // 4 M-subtiles
  const int wn   = wv >> 2;          // 2 N-halves, 2 subtiles each
  const int bm   = blockIdx.x * 64;
  const int bn   = blockIdx.y * 64;

  // ---- precompute branchless staging addresses ----
  // A: thread t loads 8 contiguous bf16: row mi = t>>2, cols kiA..kiA+7
  const int mi  = t >> 2;
  const int kiA = (t & 3) * 8;
  int mgA = bm + mi; if (mgA > M - 1) mgA = M - 1;     // clamp (guarded at store)
  long abase;
  u32  amask = 0xffffffffu;                            // zero-fill mask (shift pad)
  if (img_h > 0) {
    int h = mgA / img_w, w = mgA - h * img_w;
    int hh = h + sdy, ww = w + sdx;
    bool valid = (bm + mi < M) & (hh >= 0) & (hh < img_h) & (ww >= 0) & (ww < img_w);
    if (hh < 0) hh = 0; if (hh > img_h - 1) hh = img_h - 1;
    if (ww < 0) ww = 0; if (ww > img_w - 1) ww = img_w - 1;
    abase = (long)(hh * img_w + ww) * K + kiA;
    amask = valid ? 0xffffffffu : 0u;
  } else {
    abase = (long)mgA * K + kiA;
  }
  // B: thread t loads 8 contiguous bf16: row kiB = t>>3, cols niB..niB+7
  const int kiB = t >> 3;
  const int niB = (t & 7) * 8;
  const long bbase = (long)kiB * N + bn + niB;

  v8f acc0 = {}; v8f acc1 = {};

  for (int k0 = 0; k0 < K; k0 += 32) {
    uint4 av = *(const uint4*)(A + abase + k0);
    av.x &= amask; av.y &= amask; av.z &= amask; av.w &= amask;
    uint4 bv = *(const uint4*)(B + bbase + (long)k0 * N);
    __syncthreads();                        // prev iteration done reading LDS
    *(uint4*)&lA[mi][kiA]  = av;
    *(uint4*)&lB[kiB][niB] = bv;
    __syncthreads();

    // A fragment: lane<16 rows hold K 0..7,16..23; lane>=16 hold 8..15,24..31
    v16bf af;
    {
      int mrow = wm * 16 + (lane & 15);
      int half = lane >> 4;
      #pragma unroll
      for (int e = 0; e < 16; ++e) {
        int kk = ((e < 8) ? 0 : 16) + half * 8 + (e & 7);
        af[e] = __builtin_bit_cast(__bf16, lA[mrow][kk]);
      }
    }
    // B fragments: lane holds column N=lane%16, K in same order as A
    v16bf bf0, bf1;
    #pragma unroll
    for (int e = 0; e < 16; ++e) {
      int kb = ((e < 8) ? 0 : 16) + (lane >> 4) * 8 + (e & 7);
      bf0[e] = __builtin_bit_cast(__bf16, lB[kb][wn * 32 + (lane & 15)]);
      bf1[e] = __builtin_bit_cast(__bf16, lB[kb][wn * 32 + 16 + (lane & 15)]);
    }
    acc0 = __builtin_amdgcn_wmma_f32_16x16x32_bf16(false, af, false, bf0,
                                                   (short)0, acc0, false, false);
    acc1 = __builtin_amdgcn_wmma_f32_16x16x32_bf16(false, af, false, bf1,
                                                   (short)0, acc1, false, false);
  }

  // C/D layout: VGPR r: lane<16 -> (M=r, N=lane); lane>=16 -> (M=r+8, N=lane-16)
  const int half = lane >> 4;
  const int lmod = lane & 15;
  #pragma unroll
  for (int r = 0; r < 8; ++r) {
    int mg = bm + wm * 16 + r + half * 8;
    if (mg >= M) continue;
    #pragma unroll
    for (int s = 0; s < 2; ++s) {
      int ng = bn + wn * 32 + s * 16 + lmod;
      if (ng >= N) continue;
      float v = (s == 0) ? acc0[r] : acc1[r];
      if (bias) v += bias[ng];
      if (relu) v = fmaxf(v, 0.f);
      if (Cin)  v += Cin[(long)mg * N + ng];
      if (Cout) Cout[(long)mg * N + ng] = v;
      if (Obf)  Obf[(long)mg * obf_stride + obf_coloff + ng] = f2bf(v);
    }
  }
}

// ---------------------------------------------------------------------------
// Elementwise / layout kernels
// ---------------------------------------------------------------------------
__global__ void k_convert(const float* __restrict__ in, u16* __restrict__ out, long n) {
  for (long i = (long)blockIdx.x * 256 + threadIdx.x; i < n; i += (long)gridDim.x * 256)
    out[i] = f2bf(in[i]);
}

// in f32 [R, Cc] row-major -> out bf16 [Cc, R]
__global__ void k_tconv(const float* __restrict__ in, u16* __restrict__ out, int R, int Cc) {
  long n = (long)R * Cc;
  for (long i = (long)blockIdx.x * 256 + threadIdx.x; i < n; i += (long)gridDim.x * 256) {
    int r = (int)(i / Cc), c = (int)(i - (long)r * Cc);
    out[(long)c * R + r] = f2bf(in[i]);
  }
}

// layer_w [O=256][C=256][3][3] -> out [9][C][O]
__global__ void k_layerw(const float* __restrict__ in, u16* __restrict__ out) {
  long n = 9L * 256 * 256;
  for (long i = (long)blockIdx.x * 256 + threadIdx.x; i < n; i += (long)gridDim.x * 256) {
    int tap = (int)(i >> 16);
    int rem = (int)(i & 65535);
    int c = rem >> 8, o = rem & 255;
    out[i] = f2bf(in[(long)o * 2304 + c * 9 + tap]);
  }
}

// feat NCHW [256, nh*5, nw*5] -> A bf16 [nh*nw, 6400]; k = c*25 + a*5 + b
__global__ void k_im2col(const float* __restrict__ feat, u16* __restrict__ out,
                         int nh, int nw) {
  int W = nw * 5, H = nh * 5;
  long n = (long)nh * nw * 6400;
  for (long i = (long)blockIdx.x * 256 + threadIdx.x; i < n; i += (long)gridDim.x * 256) {
    int m = (int)(i / 6400), k = (int)(i - (long)m * 6400);
    int c = k / 25, ab = k - c * 25, a = ab / 5, b = ab - a * 5;
    int ih = m / nw, iw = m - ih * nw;
    out[i] = f2bf(feat[((long)c * H + ih * 5 + a) * W + iw * 5 + b]);
  }
}

// spatial score S[r,c] = -dist(center_r, center_c)
__global__ void k_spatial(float* __restrict__ S, int nh, int nw) {
  int N = nh * nw;
  long n = (long)N * N;
  for (long i = (long)blockIdx.x * 256 + threadIdx.x; i < n; i += (long)gridDim.x * 256) {
    int r = (int)(i / N), c = (int)(i - (long)r * N);
    float yr = (float)((r / nw) * 5), xr = (float)((r % nw) * 5);
    float yc = (float)((c / nw) * 5), xc = (float)((c % nw) * 5);
    float dy = yr - yc, dx = xr - xc;
    S[i] = -sqrtf(dy * dy + dx * dx);
  }
}

// per-row top-7 (block per row)
__global__ __launch_bounds__(256) void k_topk(const float* __restrict__ S,
                                              int* __restrict__ tidx, float* __restrict__ tval,
                                              int N, int Kk) {
  __shared__ float sv[1600];
  __shared__ float rv[256];
  __shared__ int   ri[256];
  int row = blockIdx.x, t = threadIdx.x;
  for (int i = t; i < N; i += 256) sv[i] = S[(long)row * N + i];
  __syncthreads();
  for (int j = 0; j < Kk; ++j) {
    float bv = -__builtin_inff(); int bi = N;
    for (int i = t; i < N; i += 256) {
      float v = sv[i];
      if (v > bv || (v == bv && i < bi)) { bv = v; bi = i; }
    }
    rv[t] = bv; ri[t] = bi; __syncthreads();
    for (int s = 128; s > 0; s >>= 1) {
      if (t < s) {
        if (rv[t + s] > rv[t] || (rv[t + s] == rv[t] && ri[t + s] < ri[t])) {
          rv[t] = rv[t + s]; ri[t] = ri[t + s];
        }
      }
      __syncthreads();
    }
    if (t == 0) {
      tidx[row * Kk + j] = ri[0];
      tval[row * Kk + j] = rv[0];
      sv[ri[0]] = -__builtin_inff();
    }
    __syncthreads();
  }
}

// GATv2 aggregation + ELU. Neighbors of node r: {c in topk(r) : c>r, val!=0} u {r}.
__global__ __launch_bounds__(256) void k_gatv2(
    const float* __restrict__ xl, const float* __restrict__ xr,
    const int* __restrict__ tidx, const float* __restrict__ tval,
    const float* __restrict__ att, const float* __restrict__ bias,
    float* __restrict__ outf, u16* __restrict__ outbf,
    int obf_stride, int obf_coloff, int H, int D, int Kk)
{
  __shared__ int   nbr[8];
  __shared__ int   nval;
  __shared__ float evals[8];
  __shared__ float alpha[8];
  __shared__ float red[256];
  int r = blockIdx.x, t = threadIdx.x;
  if (t == 0) {
    int cnt = 0;
    for (int j = 0; j < Kk; ++j) {
      int s = tidx[r * Kk + j]; float v = tval[r * Kk + j];
      if (s > r && v != 0.0f) nbr[cnt++] = s;
    }
    nbr[cnt++] = r;                    // self loop
    nval = cnt;
  }
  __syncthreads();
  int nn = nval;
  long HD = (long)H * D;
  for (int h = 0; h < H; ++h) {
    for (int j = 0; j < nn; ++j) {
      int s = nbr[j];
      float p = 0.f;
      for (int d = t; d < D; d += 256) {
        float m = xl[(long)s * HD + h * D + d] + xr[(long)r * HD + h * D + d];
        float lr = (m > 0.f) ? m : 0.2f * m;
        p += lr * att[h * D + d];
      }
      red[t] = p; __syncthreads();
      for (int st = 128; st > 0; st >>= 1) { if (t < st) red[t] += red[t + st]; __syncthreads(); }
      if (t == 0) evals[j] = red[0];
      __syncthreads();
    }
    if (t == 0) {
      float mx = -__builtin_inff();
      for (int j = 0; j < nn; ++j) mx = fmaxf(mx, evals[j]);
      float sum = 0.f;
      for (int j = 0; j < nn; ++j) { alpha[j] = expf(evals[j] - mx); sum += alpha[j]; }
      float inv = 1.f / (sum + 1e-16f);
      for (int j = 0; j < nn; ++j) alpha[j] *= inv;
    }
    __syncthreads();
    for (int d = t; d < D; d += 256) {
      float o = 0.f;
      for (int j = 0; j < nn; ++j) o += alpha[j] * xl[(long)nbr[j] * HD + h * D + d];
      float v = o + bias[h * D + d];
      v = (v > 0.f) ? v : (expf(v) - 1.f);     // ELU
      if (outf)  outf[(long)r * HD + h * D + d] = v;
      if (outbf) outbf[(long)r * obf_stride + obf_coloff + h * D + d] = f2bf(v);
    }
    __syncthreads();
  }
}

// convT scatter: xs = feat + relu(up), channels-last bf16 out.
// up[o, iw*5+a, ih*5+b] = U[node, o*25+a*5+b] + bias[o]  (square images)
__global__ void k_scatter_convT(const float* __restrict__ U, const float* __restrict__ cb,
                                const float* __restrict__ feat, u16* __restrict__ xs_bf,
                                int nh, int nw) {
  int W = nw * 5, H = nh * 5;
  long n = (long)nh * nw * 6400;
  for (long i = (long)blockIdx.x * 256 + threadIdx.x; i < n; i += (long)gridDim.x * 256) {
    int m = (int)(i / 6400), k = (int)(i - (long)m * 6400);
    int o = k / 25, ab = k - o * 25, a = ab / 5, b = ab - a * 5;
    int ih = m / nw, iw = m - ih * nw;
    int row = iw * 5 + a, col = ih * 5 + b;
    float v = U[i] + cb[o];
    v = fmaxf(v, 0.f);
    v += feat[((long)o * H + row) * W + col];
    xs_bf[((long)row * W + col) * 256 + o] = f2bf(v);
  }
}

// pre0 = bf16(inner0 + nearest_upsample(last))
__global__ void k_fuse_td(const float* __restrict__ inner0, const float* __restrict__ last,
                          u16* __restrict__ pre, int Hh, int Ww, int H2, int W2) {
  long n = (long)Hh * Ww * 256;
  for (long i = (long)blockIdx.x * 256 + threadIdx.x; i < n; i += (long)gridDim.x * 256) {
    int m = (int)(i >> 8), c = (int)(i & 255);
    int h = m / Ww, w = m - h * Ww;
    int m2 = ((h * H2) / Hh) * W2 + (w * W2) / Ww;
    pre[i] = f2bf(inner0[(long)m * 256 + c] + last[(long)m2 * 256 + c]);
  }
}

// channels-last f32 [HW,256] -> NCHW f32 [256,HW]
__global__ void k_store_nchw(const float* __restrict__ src, float* __restrict__ dst, int HW) {
  long n = (long)HW * 256;
  for (long i = (long)blockIdx.x * 256 + threadIdx.x; i < n; i += (long)gridDim.x * 256) {
    int m = (int)(i >> 8), c = (int)(i & 255);
    dst[(long)c * HW + m] = src[i];
  }
}

// ---------------------------------------------------------------------------
extern "C" void kernel_launch(void* const* d_in, const int* in_sizes, int n_in,
                              void* d_out, int out_size, void* d_ws, size_t ws_size,
                              hipStream_t stream) {
  (void)in_sizes; (void)n_in; (void)out_size; (void)ws_size;
  const int C = 256, H8 = 2048, KK = 7;

  const float* feat[2] = { (const float*)d_in[0], (const float*)d_in[1] };
  const float* inner_w = (const float*)d_in[2];
  const float* inner_b = (const float*)d_in[3];
  const float* layer_w = (const float*)d_in[4];
  const float* layer_b = (const float*)d_in[5];
  const float* convT_w = (const float*)d_in[6];
  const float* convT_b = (const float*)d_in[7];
  const float* conv1_w = (const float*)d_in[8];
  const float* conv1_b = (const float*)d_in[9];
  const float* fc2_w   = (const float*)d_in[10];
  const float* fc2_b   = (const float*)d_in[11];
  const float* fc3_w   = (const float*)d_in[12];
  const float* fc3_b   = (const float*)d_in[13];
  const float* g_wl[4]   = {(const float*)d_in[14],(const float*)d_in[20],(const float*)d_in[26],(const float*)d_in[32]};
  const float* g_bl[4]   = {(const float*)d_in[15],(const float*)d_in[21],(const float*)d_in[27],(const float*)d_in[33]};
  const float* g_wr[4]   = {(const float*)d_in[16],(const float*)d_in[22],(const float*)d_in[28],(const float*)d_in[34]};
  const float* g_br[4]   = {(const float*)d_in[17],(const float*)d_in[23],(const float*)d_in[29],(const float*)d_in[35]};
  const float* g_att[4]  = {(const float*)d_in[18],(const float*)d_in[24],(const float*)d_in[30],(const float*)d_in[36]};
  const float* g_bias[4] = {(const float*)d_in[19],(const float*)d_in[25],(const float*)d_in[31],(const float*)d_in[37]};

  // ---- workspace bump allocator (deterministic, recomputed each call) ----
  size_t off = 0;
  auto alloc = [&](size_t bytes) -> void* {
    void* p = (char*)d_ws + off;
    off = (off + bytes + 511) & ~(size_t)511;
    return p;
  };
  u16*   xs_bf[2]; xs_bf[0] = (u16*)alloc(40000L*256*2); xs_bf[1] = (u16*)alloc(10000L*256*2);
  u16*   wA     = (u16*)alloc(1600L*6400*2);      // im2col; reused as pre0
  u16*   wBig   = (u16*)alloc(6400L*256*2);       // conv1B / convTB
  u16*   smB    = (u16*)alloc(256L*256*2);        // fc2B / innerB
  u16*   fc3B   = (u16*)alloc(512L*256*2);
  u16*   gwl    = (u16*)alloc(2048L*256*2);
  u16*   gwr    = (u16*)alloc(2048L*256*2);
  u16*   layerB = (u16*)alloc(9L*256*256*2);
  u16*   emb1bf = (u16*)alloc(1600L*256*2);
  float* emb2f  = (float*)alloc(1600L*256*4);
  u16*   emb2bf = (u16*)alloc(1600L*256*2);
  u16*   embT   = (u16*)alloc(256L*1600*2);
  float* S      = (float*)alloc(1600L*1600*4);
  int*   tidx1  = (int*)alloc(1600L*KK*4);
  float* tval1  = (float*)alloc(1600L*KK*4);
  int*   tidx2  = (int*)alloc(1600L*KK*4);
  float* tval2  = (float*)alloc(1600L*KK*4);
  float* xlf    = (float*)alloc(1600L*2048*4);
  float* xrf    = (float*)alloc(1600L*2048*4);
  float* hbigf  = (float*)alloc(1600L*2048*4);
  u16*   hbigbf = (u16*)alloc(1600L*2048*2);
  u16*   hcat   = (u16*)alloc(1600L*512*2);
  float* h2f    = (float*)alloc(1600L*256*4);
  u16*   hcbf   = (u16*)alloc(1600L*256*2);
  float* U      = (float*)alloc(1600L*6400*4);    // reused as inner0 (same size)
  float* lastf  = (float*)alloc(10000L*256*4);
  u16*   lastbf = (u16*)alloc(10000L*256*2);
  float* tmp    = (float*)alloc(40000L*256*4);
  float* inner0f = U;       // 10.24M f32, safe reuse (U dead before FPN inner0)
  u16*   pre0bf  = wA;      // 10.24M u16, safe reuse (im2col dead before FPN)

  auto cdiv = [](long a, long b) { return (int)((a + b - 1) / b); };
  auto ewg  = [&](long n) { return dim3(cdiv(n, 256)); };
  auto gemm = [&](const u16* A, const u16* B, const float* Cin, const float* bias,
                  float* Cout, u16* Obf, int M, int N, int K,
                  int iw, int ih, int sdy, int sdx, int relu, int ost, int ocol) {
    dim3 g(cdiv(M, 64), cdiv(N, 64));
    gemm_wmma_bf16<<<g, 256, 0, stream>>>(A, B, Cin, bias, Cout, Obf, M, N, K,
                                          iw, ih, sdy, sdx, relu, ost, ocol);
  };

  // ======================= per-level GNN =======================
  for (int lv = 0; lv < 2; ++lv) {
    const int nh = (lv == 0) ? 40 : 20, nw = nh;
    const int Nn = nh * nw;
    const float* f = feat[lv];

    // weight prep
    k_tconv<<<ewg(256L*6400),256,0,stream>>>(conv1_w + (long)lv*C*6400, wBig, 256, 6400);
    k_tconv<<<ewg(256L*256),256,0,stream>>>(fc2_w + (long)lv*C*C, smB, 256, 256);
    k_tconv<<<ewg(256L*512),256,0,stream>>>(fc3_w + (long)lv*C*2*C, fc3B, 256, 512);

    // patch embedding: emb = relu(conv1(patches)); emb = relu(fc2(emb))
    k_im2col<<<ewg((long)Nn*6400),256,0,stream>>>(f, wA, nh, nw);
    gemm(wA, wBig, nullptr, conv1_b + lv*C, nullptr, emb1bf, Nn, 256, 6400, 0,0,0,0, 1, 256, 0);
    gemm(emb1bf, smB, nullptr, fc2_b + lv*C, emb2f, emb2bf, Nn, 256, 256, 0,0,0,0, 1, 256, 0);

    // feature-similarity kNN graph: S = emb @ emb.T, top-7 per row
    k_tconv<<<ewg((long)Nn*256),256,0,stream>>>(emb2f, embT, Nn, 256);
    gemm(emb2bf, embT, nullptr, nullptr, S, nullptr, Nn, Nn, 256, 0,0,0,0, 0, 0, 0);
    k_topk<<<Nn,256,0,stream>>>(S, tidx1, tval1, Nn, KK);

    // g1 (8 heads, D=256) then g2 (1 head) on feature graph
    k_convert<<<ewg(256L*2048),256,0,stream>>>(g_wl[0] + (long)lv*C*H8, gwl, 256L*2048);
    k_convert<<<ewg(256L*2048),256,0,stream>>>(g_wr[0] + (long)lv*C*H8, gwr, 256L*2048);
    gemm(emb2bf, gwl, nullptr, g_bl[0] + lv*H8, xlf, nullptr, Nn, 2048, 256, 0,0,0,0, 0, 0, 0);
    gemm(emb2bf, gwr, nullptr, g_br[0] + lv*H8, xrf, nullptr, Nn, 2048, 256, 0,0,0,0, 0, 0, 0);
    k_gatv2<<<Nn,256,0,stream>>>(xlf, xrf, tidx1, tval1, g_att[0] + (long)lv*8*C,
                                 g_bias[0] + lv*H8, hbigf, hbigbf, 2048, 0, 8, 256, KK);
    k_convert<<<ewg(2048L*256),256,0,stream>>>(g_wl[1] + (long)lv*H8*C, gwl, 2048L*256);
    k_convert<<<ewg(2048L*256),256,0,stream>>>(g_wr[1] + (long)lv*H8*C, gwr, 2048L*256);
    gemm(hbigbf, gwl, nullptr, g_bl[1] + lv*C, xlf, nullptr, Nn, 256, 2048, 0,0,0,0, 0, 0, 0);
    gemm(hbigbf, gwr, nullptr, g_br[1] + lv*C, xrf, nullptr, Nn, 256, 2048, 0,0,0,0, 0, 0, 0);
    k_gatv2<<<Nn,256,0,stream>>>(xlf, xrf, tidx1, tval1, g_att[1] + lv*C,
                                 g_bias[1] + lv*C, nullptr, hcat, 512, 0, 1, 256, KK);

    // spatial kNN graph, g3 then g4
    k_spatial<<<ewg((long)Nn*Nn),256,0,stream>>>(S, nh, nw);
    k_topk<<<Nn,256,0,stream>>>(S, tidx2, tval2, Nn, KK);
    k_convert<<<ewg(256L*2048),256,0,stream>>>(g_wl[2] + (long)lv*C*H8, gwl, 256L*2048);
    k_convert<<<ewg(256L*2048),256,0,stream>>>(g_wr[2] + (long)lv*C*H8, gwr, 256L*2048);
    gemm(emb2bf, gwl, nullptr, g_bl[2] + lv*H8, xlf, nullptr, Nn, 2048, 256, 0,0,0,0, 0, 0, 0);
    gemm(emb2bf, gwr, nullptr, g_br[2] + lv*H8, xrf, nullptr, Nn, 2048, 256, 0,0,0,0, 0, 0, 0);
    k_gatv2<<<Nn,256,0,stream>>>(xlf, xrf, tidx2, tval2, g_att[2] + (long)lv*8*C,
                                 g_bias[2] + lv*H8, hbigf, hbigbf, 2048, 0, 8, 256, KK);
    k_convert<<<ewg(2048L*256),256,0,stream>>>(g_wl[3] + (long)lv*H8*C, gwl, 2048L*256);
    k_convert<<<ewg(2048L*256),256,0,stream>>>(g_wr[3] + (long)lv*H8*C, gwr, 2048L*256);
    gemm(hbigbf, gwl, nullptr, g_bl[3] + lv*C, xlf, nullptr, Nn, 256, 2048, 0,0,0,0, 0, 0, 0);
    gemm(hbigbf, gwr, nullptr, g_br[3] + lv*C, xrf, nullptr, Nn, 256, 2048, 0,0,0,0, 0, 0, 0);
    k_gatv2<<<Nn,256,0,stream>>>(xlf, xrf, tidx2, tval2, g_att[3] + lv*C,
                                 g_bias[3] + lv*C, h2f, hcat, 512, 256, 1, 256, KK);

    // hc = relu(concat(h, h2) @ fc3.T + b) + h2
    gemm(hcat, fc3B, h2f, fc3_b + lv*C, nullptr, hcbf, Nn, 256, 512, 0,0,0,0, 1, 256, 0);

    // convT (5x5 stride 5) + bias + relu + residual  -> xs channels-last bf16
    k_convert<<<ewg(256L*6400),256,0,stream>>>(convT_w + (long)lv*C*C*25, wBig, 256L*6400);
    gemm(hcbf, wBig, nullptr, nullptr, U, nullptr, Nn, 6400, 256, 0,0,0,0, 0, 0, 0);
    k_scatter_convT<<<ewg((long)Nn*6400),256,0,stream>>>(U, convT_b + lv*C, f, xs_bf[lv], nh, nw);
  }

  // ======================= top-down FPN =======================
  float* out0 = (float*)d_out;
  float* out1 = (float*)d_out + 40000L * 256;

  // last = inner1(xs1); out1 = conv3x3(last)
  k_tconv<<<ewg(256L*256),256,0,stream>>>(inner_w + 1L*C*C, smB, 256, 256);
  gemm(xs_bf[1], smB, nullptr, inner_b + C, lastf, lastbf, 10000, 256, 256, 0,0,0,0, 0, 256, 0);
  k_layerw<<<ewg(9L*256*256),256,0,stream>>>(layer_w + 1L*C*C*9, layerB);
  for (int tap = 0; tap < 9; ++tap) {
    int dy = tap / 3 - 1, dx = tap % 3 - 1;
    gemm(lastbf, layerB + (long)tap*65536, tap ? tmp : nullptr,
         tap ? nullptr : layer_b + C, tmp, nullptr,
         10000, 256, 256, 100, 100, dy, dx, 0, 0, 0);
  }
  k_store_nchw<<<ewg(10000L*256),256,0,stream>>>(tmp, out1, 10000);

  // out0 = conv3x3(inner0(xs0) + upsample(last))
  k_tconv<<<ewg(256L*256),256,0,stream>>>(inner_w, smB, 256, 256);
  gemm(xs_bf[0], smB, nullptr, inner_b, inner0f, nullptr, 40000, 256, 256, 0,0,0,0, 0, 0, 0);
  k_fuse_td<<<ewg(40000L*256),256,0,stream>>>(inner0f, lastf, pre0bf, 200, 200, 100, 100);
  k_layerw<<<ewg(9L*256*256),256,0,stream>>>(layer_w, layerB);
  for (int tap = 0; tap < 9; ++tap) {
    int dy = tap / 3 - 1, dx = tap % 3 - 1;
    gemm(pre0bf, layerB + (long)tap*65536, tap ? tmp : nullptr,
         tap ? nullptr : layer_b, tmp, nullptr,
         40000, 256, 256, 200, 200, dy, dx, 0, 0, 0);
  }
  k_store_nchw<<<ewg(40000L*256),256,0,stream>>>(tmp, out0, 40000);
}
